// ConvNeXtLayer_86337432584772
// MI455X (gfx1250) — matrix operand order repaired and checked by reading, not compiled, not run on hardware
//
#include <hip/hip_runtime.h>
#include <cmath>

typedef __attribute__((ext_vector_type(16))) _Float16     v16h;
typedef __attribute__((ext_vector_type(8)))  float        v8f;
typedef __attribute__((ext_vector_type(4)))  unsigned int u32x4;
typedef __attribute__((ext_vector_type(2)))  unsigned int u32x2;

#define CDIM 384
#define HID  1536
#define SPHW 3136          // 56*56
#define NTOK 50176         // 16*3136

// ------------------- fp32 -> fp16 weight conversion -------------------
__global__ void cvt_weights(const float* __restrict__ w1,
                            const float* __restrict__ w2,
                            _Float16* __restrict__ w1h,
                            _Float16* __restrict__ w2h, int n) {
    int i = blockIdx.x * blockDim.x + threadIdx.x;
    if (i < n) { w1h[i] = (_Float16)w1[i]; w2h[i] = (_Float16)w2[i]; }
}

// ---- depthwise 7x7 conv + bias + LayerNorm -> f16 NHWC activations ----
// one block per token (b,h,w); 384 threads = one channel each (12 waves)
__global__ void conv_ln(const float* __restrict__ x,
                        const float* __restrict__ cw,
                        const float* __restrict__ cb,
                        const float* __restrict__ g,
                        const float* __restrict__ beta,
                        _Float16* __restrict__ xh) {
    __shared__ float red[CDIM];
    __shared__ float s_stat;
    const int tok = blockIdx.x;
    const int c   = threadIdx.x;
    const int b   = tok / SPHW, sp = tok - b * SPHW;
    const int h   = sp / 56,    w  = sp - (sp / 56) * 56;
    const float* xc = x + ((size_t)(b * CDIM + c)) * SPHW;
    float acc = cb[c];
    #pragma unroll
    for (int dy = 0; dy < 7; ++dy) {
        int yy = h + dy - 3;
        if (yy < 0 || yy >= 56) continue;
        #pragma unroll
        for (int dx = 0; dx < 7; ++dx) {
            int xx = w + dx - 3;
            if (xx < 0 || xx >= 56) continue;
            acc += xc[yy * 56 + xx] * cw[c * 49 + dy * 7 + dx];
        }
    }
    // mean
    red[c] = acc; __syncthreads();
    if (c < 128) red[c] += red[c + 128] + red[c + 256];
    __syncthreads();
    for (int s = 64; s > 0; s >>= 1) { if (c < s) red[c] += red[c + s]; __syncthreads(); }
    if (c == 0) s_stat = red[0] * (1.0f / CDIM);
    __syncthreads();
    const float mu = s_stat;
    const float d  = acc - mu;
    __syncthreads();
    // variance
    red[c] = d * d; __syncthreads();
    if (c < 128) red[c] += red[c + 128] + red[c + 256];
    __syncthreads();
    for (int s = 64; s > 0; s >>= 1) { if (c < s) red[c] += red[c + s]; __syncthreads(); }
    if (c == 0) s_stat = red[0] * (1.0f / CDIM);
    __syncthreads();
    const float v = d * rsqrtf(s_stat + 1e-6f) * g[c] + beta[c];
    xh[(size_t)tok * CDIM + c] = (_Float16)v;
}

// ----------------------- WMMA tiled GEMM ------------------------------
// C(M x N) = A(M x K) @ W(K x N); block tile 128x128, BK = 32 (one WMMA k-step)
// 256 threads = 8 waves in 4(M) x 2(N); each wave: 32x64 = 2x4 WMMA tiles.
// EPI==1: +bias, exact GELU, store f16     (GEMM1: X@w1)
// EPI==2: +bias, *layer_scale, +residual, scatter to NCHW f32 (GEMM2: H@w2)
template<int N, int K, int EPI>
__global__ __launch_bounds__(256)
void gemm_wmma(const _Float16* __restrict__ A,
               const _Float16* __restrict__ W,
               const float*    __restrict__ bias,
               const float*    __restrict__ ls,
               const float*    __restrict__ xres,
               _Float16*       __restrict__ outh,
               float*          __restrict__ outf) {
    constexpr int LDA = 40;                       // 32 + 8 halves pad (80B rows)
    __shared__ __align__(16) _Float16 As[128 * LDA];
    __shared__ __align__(16) _Float16 Bs[128 * LDA];   // stored [n][k]

    const int tid  = threadIdx.x;
    const int lane = tid & 31;
    const int wid  = tid >> 5;
    const int wm   = wid & 3;                     // wave row (32 rows each)
    const int wn   = wid >> 2;                    // wave col (64 cols each)
    const int lo   = lane & 15, hi = lane >> 4;
    const int m0   = blockIdx.y * 128;
    const int n0   = blockIdx.x * 128;

    v8f acc[2][4];
    #pragma unroll
    for (int i = 0; i < 2; ++i)
        #pragma unroll
        for (int j = 0; j < 4; ++j)
            acc[i][j] = v8f{0.f,0.f,0.f,0.f,0.f,0.f,0.f,0.f};

    const int arow = tid >> 1;                    // 0..127
    const int aseg = (tid & 1) * 16;              // 0 / 16 halves
    const int bkk0 = (tid >> 5) * 4;              // 0,4,...,28  (4 k rows)
    const int bnn  = (tid & 31) * 4;              // 0..124      (4 n cols)

    for (int k0 = 0; k0 < K; k0 += 32) {
        // stage A tile 128x32 (row major, b128 pairs)
        const _Float16* ga = A + (size_t)(m0 + arow) * K + k0 + aseg;
        *(u32x4*)&As[arow * LDA + aseg]     = *(const u32x4*)ga;
        *(u32x4*)&As[arow * LDA + aseg + 8] = *(const u32x4*)(ga + 8);
        // stage B tile 32x128 transposed into Bs[n][k]:
        // each thread: 4x4 (k x n) sub-block; coalesced b64 global reads,
        // register repack, vector b64 LDS stores (k contiguous in Bs)
        {
            union { u32x2 u; _Float16 h[4]; } t[4], s;
            #pragma unroll
            for (int r = 0; r < 4; ++r)
                t[r].u = *(const u32x2*)&W[(size_t)(k0 + bkk0 + r) * N + n0 + bnn];
            #pragma unroll
            for (int q = 0; q < 4; ++q) {
                s.h[0] = t[0].h[q]; s.h[1] = t[1].h[q];
                s.h[2] = t[2].h[q]; s.h[3] = t[3].h[q];
                *(u32x2*)&Bs[(bnn + q) * LDA + bkk0] = s.u;
            }
        }
        __syncthreads();

        // fragments per documented 16-bit A/B VGPR layouts (lane halves pick K range)
        union Frag { u32x4 u[2]; v16h v; } af[2], bf[4];
        #pragma unroll
        for (int i = 0; i < 2; ++i) {
            const _Float16* pa = &As[(wm * 32 + i * 16 + lo) * LDA + hi * 8];
            af[i].u[0] = *(const u32x4*)pa;         // K = 8*hi .. +7
            af[i].u[1] = *(const u32x4*)(pa + 16);  // K = 16+8*hi .. +7
        }
        #pragma unroll
        for (int j = 0; j < 4; ++j) {
            const _Float16* pb = &Bs[(wn * 64 + j * 16 + lo) * LDA + hi * 8];
            bf[j].u[0] = *(const u32x4*)pb;         // K = 8*hi .. +7
            bf[j].u[1] = *(const u32x4*)(pb + 16);  // K = 16+8*hi .. +7
        }
        #pragma unroll
        for (int i = 0; i < 2; ++i)
            #pragma unroll
            for (int j = 0; j < 4; ++j)
                acc[i][j] = __builtin_amdgcn_wmma_f32_16x16x32_f16(
                    false, af[i].v, false, bf[j].v,
                    (short)0, acc[i][j], false, false);
        __syncthreads();
    }

    // epilogue: C layout -> VGPR e holds (M = 8*hi + e, N = lo) within 16x16 tile
    #pragma unroll
    for (int i = 0; i < 2; ++i) {
        #pragma unroll
        for (int j = 0; j < 4; ++j) {
            const int col = n0 + wn * 64 + j * 16 + lo;
            const float bn = bias[col];
            const float sc = (EPI == 2) ? ls[col] : 0.0f;
            #pragma unroll
            for (int e = 0; e < 8; ++e) {
                const int row = m0 + wm * 32 + i * 16 + hi * 8 + e;
                const float v = acc[i][j][e] + bn;
                if (EPI == 1) {
                    const float gl = 0.5f * v * (1.0f + erff(v * 0.70710678118654752f));
                    outh[(size_t)row * N + col] = (_Float16)gl;
                } else {
                    const int b  = row / SPHW;
                    const int sp = row - b * SPHW;
                    const size_t oi = ((size_t)(b * CDIM + col)) * SPHW + sp;
                    outf[oi] = sc * v + xres[oi];
                }
            }
        }
    }
}

extern "C" void kernel_launch(void* const* d_in, const int* in_sizes, int n_in,
                              void* d_out, int out_size, void* d_ws, size_t ws_size,
                              hipStream_t stream) {
    const float* x    = (const float*)d_in[0];
    const float* cw   = (const float*)d_in[1];
    const float* cb   = (const float*)d_in[2];
    const float* lg   = (const float*)d_in[3];
    const float* lb   = (const float*)d_in[4];
    const float* w1   = (const float*)d_in[5];
    const float* b1   = (const float*)d_in[6];
    const float* w2   = (const float*)d_in[7];
    const float* b2   = (const float*)d_in[8];
    const float* ls   = (const float*)d_in[9];
    float* out = (float*)d_out;

    char* ws = (char*)d_ws;
    _Float16* Xh  = (_Float16*)(ws);                 // 50176*384*2  = 38,535,168 B
    _Float16* w1h = (_Float16*)(ws + 38535168);      // 384*1536*2   =  1,179,648 B
    _Float16* w2h = (_Float16*)(ws + 39714816);      // 1536*384*2   =  1,179,648 B
    _Float16* Hh  = (_Float16*)(ws + 40894464);      // 50176*1536*2 = 154,140,672 B

    const int NW = CDIM * HID;                       // 589824
    cvt_weights<<<(NW + 255) / 256, 256, 0, stream>>>(w1, w2, w1h, w2h, NW);

    conv_ln<<<NTOK, CDIM, 0, stream>>>(x, cw, cb, lg, lb, Xh);

    // GEMM1: X(50176x384) @ w1(384x1536) + b1 -> GELU -> H (f16)
    gemm_wmma<HID, CDIM, 1><<<dim3(HID / 128, NTOK / 128), 256, 0, stream>>>(
        Xh, w1h, b1, nullptr, nullptr, Hh, nullptr);

    // GEMM2: H(50176x1536) @ w2(1536x384) + b2 -> *ls + residual -> NCHW out
    gemm_wmma<CDIM, HID, 2><<<dim3(CDIM / 128, NTOK / 128), 256, 0, stream>>>(
        Hh, w2h, b2, ls, x, nullptr, out);
}
